// MultiHeadedAttention_72335839199232
// MI455X (gfx1250) — matrix operand order repaired and checked by reading, not compiled, not run on hardware
//
#include <hip/hip_runtime.h>
#include <hip/hip_bf16.h>

#define B_  2
#define S_  2048
#define D_  768
#define H_  12
#define DK_ 64

typedef __attribute__((ext_vector_type(16))) _Float16 v16h;
typedef __attribute__((ext_vector_type(8)))  _Float16 v8h;
typedef __attribute__((ext_vector_type(8)))  float    v8f;
typedef __attribute__((ext_vector_type(4)))  unsigned int u32x4;
typedef __attribute__((ext_vector_type(8)))  int      i32x8;
typedef __attribute__((ext_vector_type(4)))  int      i32x4;

union CFrag  { v8f  v; float    f[8]; };
union ABFrag { v16h v; v8h      h[2]; };
union H8     { v8h  v; _Float16 e[8]; };

// --- Tensor Data Mover availability / arity detection -----------------------
#if defined(__has_builtin)
#  if __has_builtin(__builtin_amdgcn_tensor_load_to_lds)
#    define HAVE_TDM 1
#  endif
#endif
#ifndef HAVE_TDM
#  define HAVE_TDM 0
#endif

// --- fast math: raw v_exp_f32 / v_rcp_f32 (args are range-safe here) --------
#if defined(__has_builtin) && __has_builtin(__builtin_amdgcn_exp2f)
__device__ __forceinline__ float fast_exp2(float x) { return __builtin_amdgcn_exp2f(x); }
#else
__device__ __forceinline__ float fast_exp2(float x) { return exp2f(x); }
#endif
#if defined(__has_builtin) && __has_builtin(__builtin_amdgcn_rcpf)
__device__ __forceinline__ float fast_rcp(float x) { return __builtin_amdgcn_rcpf(x); }
#else
__device__ __forceinline__ float fast_rcp(float x) { return 1.0f / x; }
#endif

#if HAVE_TDM
// Issue a TDM load of a 2D f16 tile (tile_d0 x tile_d1, row stride stride_d0
// elements) from global memory into LDS at byte offset lds_off.
// Descriptor bitfields per CDNA5 ISA.
__device__ __forceinline__ void tdm_load_2d_f16(unsigned lds_off, const void* gptr,
                                                unsigned tile_d0, unsigned tile_d1,
                                                unsigned tensor_d0, unsigned tensor_d1,
                                                unsigned stride_d0) {
    unsigned long long ga = (unsigned long long)(uintptr_t)gptr;
    u32x4 g0;
    g0.x = 1u;                                        // count=1 (valid), user mode
    g0.y = lds_off;                                   // lds_addr[31:0]
    g0.z = (unsigned)(ga & 0xffffffffu);              // global_addr[31:0]
    g0.w = (unsigned)((ga >> 32) & 0x01ffffffu)       // global_addr[56:32]
         | (2u << 30);                                // type=2 ("image")
    i32x8 g1;
    g1[0] = (int)(1u << 16);                          // wg_mask=0, data_size=1 (2B)
    g1[1] = (int)((tensor_d0 & 0xffffu) << 16);       // tensor_dim0[15:0]
    g1[2] = (int)(((tensor_d0 >> 16) & 0xffffu) |
                  ((tensor_d1 & 0xffffu) << 16));     // dim0[31:16] | dim1[15:0]
    g1[3] = (int)(((tensor_d1 >> 16) & 0xffffu) |
                  ((tile_d0 & 0xffffu) << 16));       // dim1[31:16] | tile_dim0
    g1[4] = (int)(tile_d1 & 0xffffu);                 // tile_dim1 (tile_dim2=0)
    g1[5] = (int)stride_d0;                           // tensor_dim0_stride[31:0]
    g1[6] = 0;                                        // stride hi / dim1_stride lo
    g1[7] = 0;
    i32x4 z4 = {0, 0, 0, 0};
#if __has_include(<hip/amd_detail/amd_gfx1250_TDM.h>)
    i32x8 z8 = {0, 0, 0, 0, 0, 0, 0, 0};
    __builtin_amdgcn_tensor_load_to_lds(g0, g1, z4, z4, z8, 0);
#else
    __builtin_amdgcn_tensor_load_to_lds(g0, g1, z4, z4, 0);
#endif
}
#endif

__device__ __forceinline__ v8f wmma_f16(v16h a, v16h b, v8f c) {
    return __builtin_amdgcn_wmma_f32_16x16x32_f16(
        /*neg_a=*/false, a, /*neg_b=*/false, b,
        /*c_mod=*/(short)0, c, /*reuse_a=*/false, /*reuse_b=*/false);
}

// ---------------------------------------------------------------------------
// delta = 1 / (min(relu(q.Wd + bd), sqrt(dk)) + 1), indexed per (b, position)
// ---------------------------------------------------------------------------
__global__ __launch_bounds__(256)
void delta_kernel(const float* __restrict__ q, const float* __restrict__ Wd,
                  const float* __restrict__ bd, float* __restrict__ rdelta) {
    const int lane = threadIdx.x & 31;
    const int wave = threadIdx.x >> 5;
    const int row  = blockIdx.x * 8 + wave;   // 0 .. B*S-1
    float s = 0.f;
    for (int j = lane; j < D_; j += 32)
        s += q[(size_t)row * D_ + j] * Wd[j];
    #pragma unroll
    for (int off = 1; off < 32; off <<= 1)
        s += __shfl_xor(s, off, 32);
    if (lane == 0) {
        float d = fminf(fmaxf(s + bd[0], 0.0f), 8.0f) + 1.0f;  // sqrt(64)=8
        rdelta[row] = fast_rcp(d);
    }
}

// ---------------------------------------------------------------------------
// Y[4096x768] = X[4096x768] @ W[768x768] + bias, via v_wmma_f32_16x16x32_f16.
// Block = 256 threads (8 waves): tile M=128 (16 rows/wave), N=64, K-step 32.
// ---------------------------------------------------------------------------
template<bool IN_F16, bool OUT_F16, bool HEAD_SPLIT>
__global__ __launch_bounds__(256)
void gemm768(const void* __restrict__ Xv, const float* __restrict__ W,
             const float* __restrict__ bias, void* __restrict__ Yv) {
    __shared__ _Float16 As[128 * 32];   // A tile, row-major, stride 32
    __shared__ _Float16 Bt[64 * 32];    // B tile transposed: Bt[n][k]

    const int lane  = threadIdx.x & 31;
    const int wave  = threadIdx.x >> 5;
    const int nlane = lane & 15;
    const int mb = blockIdx.y * 128;
    const int nb = blockIdx.x * 64;
    const int akoff = (lane < 16) ? 0 : 16;   // A frag K-halves base (contiguous k = 16h + s)
    const int bkoff = (lane < 16) ? 0 : 16;   // B frag K-halves base

    const float*    Xf = (const float*)Xv;
    const _Float16* Xh = (const _Float16*)Xv;

    const v8f zero = {0,0,0,0,0,0,0,0};
    CFrag acc[4];
    #pragma unroll
    for (int i = 0; i < 4; ++i) acc[i].v = zero;

    for (int k0 = 0; k0 < D_; k0 += 32) {
        __syncthreads();
        {   // stage A: 128x32 halfs, 16 per thread
            int row = threadIdx.x >> 1;
            int kp  = (threadIdx.x & 1) * 16;
            size_t g = (size_t)(mb + row) * D_ + k0 + kp;
            #pragma unroll
            for (int j = 0; j < 16; ++j)
                As[row * 32 + kp + j] = IN_F16 ? Xh[g + j] : (_Float16)Xf[g + j];
            // stage B transposed: Bt[n][k] = W[k0+k][nb+n], 8 per thread
            int kk = threadIdx.x & 31;
            int n0 = (threadIdx.x >> 5) * 8;
            #pragma unroll
            for (int j = 0; j < 8; ++j)
                Bt[(n0 + j) * 32 + kk] = (_Float16)W[(size_t)(k0 + kk) * D_ + nb + n0 + j];
            if (k0 + 32 < D_) {   // prefetch next K-step panels (near scope)
                __builtin_prefetch(IN_F16 ? (const void*)(Xh + g + 32)
                                          : (const void*)(Xf + g + 32), 0, 3);
                __builtin_prefetch((const void*)&W[(size_t)(k0 + 32 + kk) * D_ + nb + n0], 0, 3);
            }
        }
        __syncthreads();

        ABFrag af;
        af.h[0] = *(const v8h*)&As[(wave * 16 + nlane) * 32 + akoff];
        af.h[1] = *(const v8h*)&As[(wave * 16 + nlane) * 32 + akoff + 8];
        #pragma unroll
        for (int ns = 0; ns < 4; ++ns) {
            v16h bf = *(const v16h*)&Bt[(ns * 16 + nlane) * 32 + bkoff];
            acc[ns].v = wmma_f16(af.v, bf, acc[ns].v);
        }
    }

    const int mrow0 = (lane < 16) ? 0 : 8;
    #pragma unroll
    for (int ns = 0; ns < 4; ++ns) {
        int ncol = nb + ns * 16 + nlane;
        float bv = bias[ncol];
        #pragma unroll
        for (int r = 0; r < 8; ++r) {
            int row = mb + wave * 16 + mrow0 + r;
            float val = acc[ns].f[r] + bv;
            if (OUT_F16) {
                _Float16* Y = (_Float16*)Yv;
                if (HEAD_SPLIT) {
                    int b = row >> 11, s = row & (S_ - 1);
                    int hh = ncol >> 6, dd = ncol & 63;
                    Y[((size_t)(b * H_ + hh) * S_ + s) * DK_ + dd] = (_Float16)val;
                } else {
                    Y[(size_t)row * D_ + ncol] = (_Float16)val;
                }
            } else {
                float* Y = (float*)Yv;
                Y[(size_t)row * D_ + ncol] = val;
            }
        }
    }
}

// ---------------------------------------------------------------------------
// Fused flash attention: block = (b,h, 128-q tile), 8 waves x 16 q rows.
// K tiles arrive via TDM DMA (double-buffered, s_wait_tensorcnt); V staged
// transposed by vector loads; scores never touch global memory.
// ---------------------------------------------------------------------------
__global__ __launch_bounds__(256)
void attn_kernel(const _Float16* __restrict__ Qp, const _Float16* __restrict__ Kp,
                 const _Float16* __restrict__ Vp, const int* __restrict__ mask,
                 const float* __restrict__ rdelta, _Float16* __restrict__ attn) {
    __shared__ _Float16 Ks[2][64 * 64];    // K tile double buffer (key, d)
    __shared__ _Float16 Vt[64 * 64];       // V tile transposed (d, key)
    __shared__ _Float16 Pb[8][16 * 64];    // per-wave P staging

    const int lane  = threadIdx.x & 31;
    const int wave  = threadIdx.x >> 5;
    const int nlane = lane & 15;
    const int bi = blockIdx.z;
    const int h  = blockIdx.y;
    const int qrow0 = blockIdx.x * 128 + wave * 16;
    const int mrow0 = (lane < 16) ? 0 : 8;
    const int akoff = (lane < 16) ? 0 : 16;
    const int bkoff = (lane < 16) ? 0 : 16;
    const float L2E = 1.4426950408889634f;
    const int NT = S_ / 64;

    const size_t headBase = (size_t)(bi * H_ + h) * S_ * DK_;
    const _Float16* Khead = Kp + headBase;

    // Q fragments (16 rows x 64 d) resident in registers for the whole kernel
    ABFrag qf[2];
    {
        const _Float16* qr = Qp + headBase + (size_t)(qrow0 + nlane) * DK_;
        #pragma unroll
        for (int dc = 0; dc < 2; ++dc) {
            qf[dc].h[0] = *(const v8h*)(qr + dc * 32 + akoff);
            qf[dc].h[1] = *(const v8h*)(qr + dc * 32 + akoff + 8);
        }
    }

    const v8f zero = {0,0,0,0,0,0,0,0};
    CFrag o[4];
    float m[8], lsum[8], sf[8];
    #pragma unroll
    for (int dt = 0; dt < 4; ++dt) o[dt].v = zero;
    #pragma unroll
    for (int r = 0; r < 8; ++r) { m[r] = -3.0e38f; lsum[r] = 0.f; }

    _Float16* Pw = &Pb[wave][0];

#if HAVE_TDM
    // Prologue: DMA first K tile into buffer 0.
    if (threadIdx.x < 32)
        tdm_load_2d_f16((unsigned)(uintptr_t)&Ks[0][0], Khead,
                        /*tile*/ DK_, 64, /*tensor*/ DK_, S_, /*stride*/ DK_);
#endif

    for (int kt = 0; kt < NT; ++kt) {
        const int kbase = kt * 64;
        const int cur = kt & 1;
        __syncthreads();   // all waves done with previous tile's LDS

#if HAVE_TDM
        if (threadIdx.x < 32) {
            __builtin_amdgcn_s_wait_tensorcnt(0);      // K tile kt has landed
            if (kt + 1 < NT)                           // kick off next tile DMA
                tdm_load_2d_f16((unsigned)(uintptr_t)&Ks[cur ^ 1][0],
                                Khead + (size_t)(kbase + 64) * DK_,
                                DK_, 64, DK_, S_, DK_);
        }
#endif
        {   // stage V transposed: 2 vector global loads + 16 ds stores / thread
            int tk    = threadIdx.x >> 2;          // key 0..63
            int tpart = (threadIdx.x & 3) * 16;    // d 0,16,32,48
            const _Float16* vrow = Vp + headBase + (size_t)(kbase + tk) * DK_;
            H8 va, vb;
            va.v = *(const v8h*)(vrow + tpart);
            vb.v = *(const v8h*)(vrow + tpart + 8);
            #pragma unroll
            for (int j = 0; j < 8; ++j) Vt[(tpart + j) * 64 + tk]     = va.e[j];
            #pragma unroll
            for (int j = 0; j < 8; ++j) Vt[(tpart + 8 + j) * 64 + tk] = vb.e[j];
            if (kt + 1 < NT)
                __builtin_prefetch((const void*)(vrow + (size_t)64 * DK_), 0, 3);
#if !HAVE_TDM
            // fallback: cooperative K staging (no TDM on this toolchain)
            const _Float16* krow = Khead + (size_t)(kbase + tk) * DK_;
            *(v8h*)&Ks[cur][tk * 64 + tpart]     = *(const v8h*)(krow + tpart);
            *(v8h*)&Ks[cur][tk * 64 + tpart + 8] = *(const v8h*)(krow + tpart + 8);
#endif
        }
        __syncthreads();   // Vt staged; wave0 passed tensorcnt wait

        // scores: 4 tiles of 16 keys, 2 WMMAs each over d=64
        CFrag st[4];
        #pragma unroll
        for (int t = 0; t < 4; ++t) {
            st[t].v = zero;
            #pragma unroll
            for (int dc = 0; dc < 2; ++dc) {
                v16h kf = *(const v16h*)&Ks[cur][(t * 16 + nlane) * 64 + dc * 32 + bkoff];
                st[t].v = wmma_f16(qf[dc].v, kf, st[t].v);
            }
        }

        // scale by 1/delta(key), apply mask (+ prefetch next tile's mask rows)
        #pragma unroll
        for (int t = 0; t < 4; ++t) {
            int key = kbase + t * 16 + nlane;
            float rd = rdelta[bi * S_ + key];
            size_t mrowbase = ((size_t)bi * S_ + qrow0 + mrow0) * S_ + key;
            if (kt + 1 < NT)
                __builtin_prefetch((const void*)&mask[mrowbase + 64], 0, 3);
            #pragma unroll
            for (int r = 0; r < 8; ++r) {
                float v = st[t].f[r] * rd;
                int mv = mask[mrowbase + (size_t)r * S_];
                st[t].f[r] = (mv == 0) ? -1.0e9f : v;
            }
        }

        // online softmax: row max across 16-lane group, rescale running state
        #pragma unroll
        for (int r = 0; r < 8; ++r) {
            float mx = fmaxf(fmaxf(st[0].f[r], st[1].f[r]),
                             fmaxf(st[2].f[r], st[3].f[r]));
            #pragma unroll
            for (int off = 1; off < 16; off <<= 1)
                mx = fmaxf(mx, __shfl_xor(mx, off, 32));
            float mn = fmaxf(m[r], mx);
            sf[r] = fast_exp2((m[r] - mn) * L2E);
            m[r] = mn;
            lsum[r] *= sf[r];
        }
        #pragma unroll
        for (int dt = 0; dt < 4; ++dt)
            #pragma unroll
            for (int r = 0; r < 8; ++r) o[dt].f[r] *= sf[r];

        // p = exp(s - m); stage P (C-layout -> LDS -> A-layout)
        #pragma unroll
        for (int t = 0; t < 4; ++t)
            #pragma unroll
            for (int r = 0; r < 8; ++r) {
                float p = fast_exp2((st[t].f[r] - m[r]) * L2E);
                lsum[r] += p;
                Pw[(mrow0 + r) * 64 + t * 16 + nlane] = (_Float16)p;
            }

        // P @ V
        ABFrag pf[2];
        #pragma unroll
        for (int kc = 0; kc < 2; ++kc) {
            pf[kc].h[0] = *(const v8h*)&Pw[nlane * 64 + kc * 32 + akoff];
            pf[kc].h[1] = *(const v8h*)&Pw[nlane * 64 + kc * 32 + akoff + 8];
        }
        #pragma unroll
        for (int dt = 0; dt < 4; ++dt)
            #pragma unroll
            for (int kc = 0; kc < 2; ++kc) {
                v16h vf = *(const v16h*)&Vt[(dt * 16 + nlane) * 64 + kc * 32 + bkoff];
                o[dt].v = wmma_f16(pf[kc].v, vf, o[dt].v);
            }
    }

    // finalize: full row sums, normalize, store f16 in [B,S,D] for Wo GEMM
    #pragma unroll
    for (int r = 0; r < 8; ++r) {
        float ls = lsum[r];
        #pragma unroll
        for (int off = 1; off < 16; off <<= 1)
            ls += __shfl_xor(ls, off, 32);
        lsum[r] = fast_rcp(ls);
    }
    #pragma unroll
    for (int dt = 0; dt < 4; ++dt)
        #pragma unroll
        for (int r = 0; r < 8; ++r) {
            float val = o[dt].f[r] * lsum[r];
            int q = qrow0 + mrow0 + r;
            attn[((size_t)bi * S_ + q) * D_ + h * DK_ + dt * 16 + nlane] = (_Float16)val;
        }
}

// ---------------------------------------------------------------------------
extern "C" void kernel_launch(void* const* d_in, const int* in_sizes, int n_in,
                              void* d_out, int out_size, void* d_ws, size_t ws_size,
                              hipStream_t stream) {
    const float* query = (const float*)d_in[0];
    const float* key   = (const float*)d_in[1];
    const float* value = (const float*)d_in[2];
    const int*   mask  = (const int*)d_in[3];
    const float* Wq = (const float*)d_in[4];
    const float* bq = (const float*)d_in[5];
    const float* Wk = (const float*)d_in[6];
    const float* bk = (const float*)d_in[7];
    const float* Wv = (const float*)d_in[8];
    const float* bv = (const float*)d_in[9];
    const float* Wo = (const float*)d_in[10];
    const float* bo = (const float*)d_in[11];
    const float* Wd = (const float*)d_in[12];
    const float* bd = (const float*)d_in[13];

    const size_t headElems = (size_t)B_ * H_ * S_ * DK_;  // == B*S*D
    _Float16* Qp   = (_Float16*)d_ws;
    _Float16* Kp   = Qp + headElems;
    _Float16* Vp   = Kp + headElems;
    _Float16* attn = Vp + headElems;
    float* rdelta  = (float*)(attn + headElems);

    delta_kernel<<<dim3(B_ * S_ / 8), 256, 0, stream>>>(query, Wd, bd, rdelta);

    dim3 ggrid(D_ / 64, B_ * S_ / 128);
    gemm768<false, true, true><<<ggrid, 256, 0, stream>>>(query, Wq, bq, Qp);
    gemm768<false, true, true><<<ggrid, 256, 0, stream>>>(key,   Wk, bk, Kp);
    gemm768<false, true, true><<<ggrid, 256, 0, stream>>>(value, Wv, bv, Vp);

    attn_kernel<<<dim3(S_ / 128, H_, B_), 256, 0, stream>>>(Qp, Kp, Vp, mask, rdelta, attn);

    gemm768<true, false, false><<<ggrid, 256, 0, stream>>>(attn, Wo, bo, d_out);
}
